// GRUMT_85590108274912
// MI455X (gfx1250) — matrix operand order repaired and checked by reading, not compiled, not run on hardware
//
#include <hip/hip_runtime.h>
#include <math.h>

#define H_    256
#define L_    512
#define VOUT  50000
#define NBLK  64
#define TPB   256
#define NTILES (VOUT / 16)          // 3125 (exact)
#define NWAVES (NBLK * 8)           // 512 waves total
#define EOS_TOK 1

// ---- workspace layout (32-bit word offsets) ----
#define WS_CNT   0
#define WS_GEN   1
#define WS_TOK   4
#define WS_DONE  5
#define WS_LOSS  6
#define WS_YTGT  7
#define WS_H2BF  16                      // 256 bf16 = 128 words (64B aligned)
#define WS_PART  160                     // 64 blocks * 4 words
#define WS_ENC   512                     // 512*256 f32 = 131072 words
#define WS_WBF   131584                  // 50000*256 bf16 = 6,400,000 words
#define WS_WORDS_FULL (131584 + (VOUT / 2) * H_)

typedef unsigned short ushort_t;
typedef __attribute__((ext_vector_type(8)))  float   f32x8;
typedef __attribute__((ext_vector_type(8)))  __bf16  bf16x8;
typedef __attribute__((ext_vector_type(16))) __bf16  bf16x16;

union Afrag { bf16x16 v; bf16x8 h[2]; };

struct Params {
  const int *in_tok, *in_len, *tgt_tok, *tgt_len;
  const float *enc_emb, *dec_emb;
  const float *enc_W_ih, *enc_W_hh, *enc_b_ih, *enc_b_hh;
  const float *dec_W_ih, *dec_W_hh, *dec_b_ih, *dec_b_hh;
  const float *attn_W, *attn_b, *comb_W, *comb_b, *out_W, *out_b;
  float* out;
  unsigned* ws;
  int useWbf;
};

__device__ __forceinline__ ushort_t f32_to_bf16(float f) {
  unsigned u = __float_as_uint(f);
  return (ushort_t)((u + 0x7fffu + ((u >> 16) & 1u)) >> 16);   // RNE
}

__device__ __forceinline__ bf16x8 cvt8(const float* p) {
  union { bf16x8 v; ushort_t u[8]; } r;
  #pragma unroll
  for (int j = 0; j < 8; ++j) r.u[j] = f32_to_bf16(p[j]);
  return r.v;
}

__device__ __forceinline__ float sigm(float x) { return 1.0f / (1.0f + __expf(-x)); }

// Load the 8 A fragments (16x32 bf16 each) of one 16-row tile.
// A layout: lanes 0-15 carry M=0..15 with K 0-7 (vgpr0-3) / 16-23 (vgpr4-7);
// lanes 16-31 carry M=0..15 with K 8-15 / 24-31.
__device__ __forceinline__ void load_a_bf16(Afrag* a, const ushort_t* pw, int hi) {
  #pragma unroll
  for (int c = 0; c < 8; ++c) {
    a[c].h[0] = *(const bf16x8*)(pw + c * 32 + hi * 8);
    a[c].h[1] = *(const bf16x8*)(pw + c * 32 + 16 + hi * 8);
  }
}

__device__ __forceinline__ void load_a_f32(Afrag* a, const float* pw, int hi) {
  #pragma unroll
  for (int c = 0; c < 8; ++c) {
    a[c].h[0] = cvt8(pw + c * 32 + hi * 8);
    a[c].h[1] = cvt8(pw + c * 32 + 16 + hi * 8);
  }
}

__device__ __forceinline__ f32x8 wmma8(const Afrag* a, const bf16x16* Bf) {
  f32x8 acc = {};
  #pragma unroll
  for (int c = 0; c < 8; ++c)
    acc = __builtin_amdgcn_wmma_f32_16x16x32_bf16(
        false, a[c].v, false, Bf[c], (short)0, acc, false, false);
  return acc;
}

// grid-wide generation barrier (all 64 blocks co-resident)
__device__ void grid_sync(unsigned* ws) {
  __threadfence();
  __syncthreads();
  if (threadIdx.x == 0) {
    unsigned g = __hip_atomic_load(&ws[WS_GEN], __ATOMIC_RELAXED, __HIP_MEMORY_SCOPE_AGENT);
    unsigned a = __hip_atomic_fetch_add(&ws[WS_CNT], 1u, __ATOMIC_ACQ_REL, __HIP_MEMORY_SCOPE_AGENT);
    if (a == (unsigned)(gridDim.x - 1)) {
      __hip_atomic_store(&ws[WS_CNT], 0u, __ATOMIC_RELAXED, __HIP_MEMORY_SCOPE_AGENT);
      __hip_atomic_fetch_add(&ws[WS_GEN], 1u, __ATOMIC_RELEASE, __HIP_MEMORY_SCOPE_AGENT);
    } else {
      while (__hip_atomic_load(&ws[WS_GEN], __ATOMIC_ACQUIRE, __HIP_MEMORY_SCOPE_AGENT) == g)
        __builtin_amdgcn_s_sleep(2);
    }
  }
  __syncthreads();
  __threadfence();
}

// one GRU cell: thread i produces h2[i]; x,h,h2 in LDS
__device__ void gru_cell(const float* x, const float* h,
                         const float* Wih, const float* Whh,
                         const float* bih, const float* bhh, float* h2) {
  int i = threadIdx.x;
  float gir = bih[i], giz = bih[H_ + i], gin = bih[2 * H_ + i];
  float ghr = bhh[i], ghz = bhh[H_ + i], ghn = bhh[2 * H_ + i];
  const float* wi0 = Wih + (size_t)i * H_;
  const float* wi1 = Wih + (size_t)(H_ + i) * H_;
  const float* wi2 = Wih + (size_t)(2 * H_ + i) * H_;
  const float* wh0 = Whh + (size_t)i * H_;
  const float* wh1 = Whh + (size_t)(H_ + i) * H_;
  const float* wh2 = Whh + (size_t)(2 * H_ + i) * H_;
  for (int k = 0; k < H_; ++k) {
    float xk = x[k], hk = h[k];
    gir = fmaf(wi0[k], xk, gir); giz = fmaf(wi1[k], xk, giz); gin = fmaf(wi2[k], xk, gin);
    ghr = fmaf(wh0[k], hk, ghr); ghz = fmaf(wh1[k], hk, ghz); ghn = fmaf(wh2[k], hk, ghn);
  }
  float r = sigm(gir + ghr);
  float z = sigm(giz + ghz);
  float n = tanhf(gin + r * ghn);
  h2[i] = (1.0f - z) * n + z * h[i];
}

__global__ void grumt_init(unsigned* ws) {
  if (threadIdx.x < 16) ws[threadIdx.x] = 0u;
}

__global__ __launch_bounds__(TPB, 1) void grumt_main(Params p) {
  __shared__ float sh_x[H_], sh_h[H_], sh_h2[H_], sh_c[H_], sh_o[H_];
  __shared__ float sh_a[L_], sh_red[TPB];
  __shared__ float redM[16], redS[16];
  __shared__ int   redI[16];
  __shared__ int   sh_flag;

  unsigned* wsU = p.ws;
  float*    wsF = (float*)p.ws;
  int*      wsI = (int*)p.ws;
  float*    wsEnc = wsF + WS_ENC;
  ushort_t* wsH2b = (ushort_t*)(wsU + WS_H2BF);
  ushort_t* wsWb  = (ushort_t*)(wsU + WS_WBF);

  const int tid = threadIdx.x, bid = blockIdx.x;
  const int wave = tid >> 5, lane = tid & 31;
  const int hi = lane >> 4, m16 = lane & 15;

  // ---------------- Phase 0: encoder (block 0) || out_W -> bf16 (blocks 1..63)
  if (bid == 0) {
    const int inLen = p.in_len[0];
    sh_h[tid] = 0.0f;
    __syncthreads();
    for (int t = 0; t < inLen; ++t) {
      int tok = p.in_tok[t];
      sh_x[tid] = p.enc_emb[(size_t)tok * H_ + tid];
      __syncthreads();
      gru_cell(sh_x, sh_h, p.enc_W_ih, p.enc_W_hh, p.enc_b_ih, p.enc_b_hh, sh_h2);
      __syncthreads();
      sh_h[tid] = sh_h2[tid];
      wsEnc[(size_t)t * H_ + tid] = sh_h2[tid];
      __syncthreads();
    }
    for (int t = inLen; t < L_; ++t) wsEnc[(size_t)t * H_ + tid] = 0.0f;
  } else if (p.useWbf) {
    const size_t total = (size_t)VOUT * H_;
    size_t idx = (size_t)(bid - 1) * TPB + tid;
    const size_t stride = (size_t)(NBLK - 1) * TPB;
    for (; idx < total; idx += stride) wsWb[idx] = f32_to_bf16(p.out_W[idx]);
  }
  grid_sync(wsU);

  const int T = p.tgt_len[0];
  for (int t = 0; t < T; ++t) {
    const int tgt = p.tgt_tok[t];

    // ---------------- Phase A (block 0): attention + combine + GRU -> h2
    if (bid == 0) {
      int tok = wsI[WS_TOK];
      sh_x[tid] = p.dec_emb[(size_t)tok * H_ + tid];
      __syncthreads();
      // attn logits: rows tid and tid+256 of attn_W (512 x 512), input [emb; h]
      #pragma unroll
      for (int rr = 0; rr < 2; ++rr) {
        int r = tid + rr * H_;
        const float* wr = p.attn_W + (size_t)r * (2 * H_);
        float acc = p.attn_b[r];
        for (int j = 0; j < H_; ++j) acc = fmaf(wr[j], sh_x[j], acc);
        for (int j = 0; j < H_; ++j) acc = fmaf(wr[H_ + j], sh_h[j], acc);
        sh_a[r] = acc;
      }
      __syncthreads();
      // softmax over 512
      sh_red[tid] = fmaxf(sh_a[tid], sh_a[tid + H_]);
      __syncthreads();
      for (int s = TPB / 2; s > 0; s >>= 1) {
        if (tid < s) sh_red[tid] = fmaxf(sh_red[tid], sh_red[tid + s]);
        __syncthreads();
      }
      float amx = sh_red[0];
      __syncthreads();
      float e0 = __expf(sh_a[tid] - amx), e1 = __expf(sh_a[tid + H_] - amx);
      sh_red[tid] = e0 + e1;
      __syncthreads();
      for (int s = TPB / 2; s > 0; s >>= 1) {
        if (tid < s) sh_red[tid] += sh_red[tid + s];
        __syncthreads();
      }
      float asum = sh_red[0];
      __syncthreads();
      sh_a[tid] = e0 / asum; sh_a[tid + H_] = e1 / asum;
      __syncthreads();
      // ctx = attn @ encoder_outputs  (coalesced over tid)
      float cx = 0.0f;
      for (int l = 0; l < L_; ++l) cx = fmaf(sh_a[l], wsEnc[(size_t)l * H_ + tid], cx);
      sh_c[tid] = cx;
      __syncthreads();
      // o = relu(comb_W @ [emb; ctx] + comb_b)
      {
        const float* wr = p.comb_W + (size_t)tid * (2 * H_);
        float acc = p.comb_b[tid];
        for (int j = 0; j < H_; ++j) acc = fmaf(wr[j], sh_x[j], acc);
        for (int j = 0; j < H_; ++j) acc = fmaf(wr[H_ + j], sh_c[j], acc);
        sh_o[tid] = fmaxf(acc, 0.0f);
      }
      __syncthreads();
      gru_cell(sh_o, sh_h, p.dec_W_ih, p.dec_W_hh, p.dec_b_ih, p.dec_b_hh, sh_h2);
      __syncthreads();
      wsH2b[tid] = f32_to_bf16(sh_h2[tid]);   // publish h2 (bf16) for the GEMV
    }
    grid_sync(wsU);

    // ---------------- Phase B (all blocks): y = out_W @ h2 via WMMA, online softmax stats
    {
      // B fragments: broadcast h2 into every column. Same K order as A:
      // lanes 0-15 hold K 0-7 / 16-23, lanes 16-31 hold K 8-15 / 24-31.
      bf16x16 Bf[8];
      #pragma unroll
      for (int c = 0; c < 8; ++c) {
        Afrag bt;
        bt.h[0] = *(const bf16x8*)(wsH2b + c * 32 + hi * 8);
        bt.h[1] = *(const bf16x8*)(wsH2b + c * 32 + 16 + hi * 8);
        Bf[c] = bt.v;
      }

      float mmax = -INFINITY, ssum = 0.0f;
      int   amax = 0;
      const int gw = bid * 8 + wave;

      // online log-sum-exp / argmax update over 8 rows of one tile
      auto extract = [&](const f32x8& acc, int r0) {
        if (m16 == 0) {
          const int rb = r0 + hi * 8;
          #pragma unroll
          for (int j = 0; j < 8; ++j) {
            int r2 = rb + j;
            float val = acc[j] + p.out_b[r2];
            if (r2 == tgt) wsF[WS_YTGT] = val;
            if (val > mmax) { ssum = ssum * __expf(mmax - val) + 1.0f; mmax = val; amax = r2; }
            else             ssum += __expf(val - mmax);
          }
        }
      };

      if (p.useWbf) {
        // software-pipelined: load A frags of tile t+1 while WMMAs of tile t run
        Afrag a0[8], a1[8];
        int tile = gw;
        load_a_bf16(a0, wsWb + (size_t)(tile * 16 + m16) * H_, hi);
        while (true) {
          int t1 = tile + NWAVES;
          if (t1 < NTILES) {
            load_a_bf16(a1, wsWb + (size_t)(t1 * 16 + m16) * H_, hi);
            if (t1 + NWAVES < NTILES)
              __builtin_prefetch(wsWb + (size_t)((t1 + NWAVES) * 16 + m16) * H_, 0, 1);
          }
          f32x8 acc = wmma8(a0, Bf);
          extract(acc, tile * 16);
          if (t1 >= NTILES) break;
          tile = t1;
          int t2 = tile + NWAVES;
          if (t2 < NTILES) {
            load_a_bf16(a0, wsWb + (size_t)(t2 * 16 + m16) * H_, hi);
            if (t2 + NWAVES < NTILES)
              __builtin_prefetch(wsWb + (size_t)((t2 + NWAVES) * 16 + m16) * H_, 0, 1);
          }
          acc = wmma8(a1, Bf);
          extract(acc, tile * 16);
          if (t2 >= NTILES) break;
          tile = t2;
        }
      } else {
        // fallback: stream f32 out_W, convert in-register (same WMMA path, 2x L2 bytes)
        Afrag a[8];
        for (int tile = gw; tile < NTILES; tile += NWAVES) {
          load_a_f32(a, p.out_W + (size_t)(tile * 16 + m16) * H_, hi);
          if (tile + NWAVES < NTILES)
            __builtin_prefetch(p.out_W + (size_t)((tile + NWAVES) * 16 + m16) * H_, 0, 1);
          f32x8 acc = wmma8(a, Bf);
          extract(acc, tile * 16);
        }
      }

      if (m16 == 0) { int slot = wave * 2 + hi; redM[slot] = mmax; redS[slot] = ssum; redI[slot] = amax; }
      __syncthreads();
      if (tid == 0) {
        float bm = -INFINITY, bs = 0.0f; int bi = 0x7fffffff;
        for (int s2 = 0; s2 < 16; ++s2) {
          float m2 = redM[s2];
          if (m2 > bm) { bm = m2; bi = redI[s2]; }
          else if (m2 == bm && redI[s2] < bi) bi = redI[s2];
        }
        for (int s2 = 0; s2 < 16; ++s2) bs += redS[s2] * __expf(redM[s2] - bm);
        wsF[WS_PART + bid * 4 + 0] = bm;
        wsF[WS_PART + bid * 4 + 1] = bs;
        wsI[WS_PART + bid * 4 + 2] = bi;
      }
      __syncthreads();
    }
    grid_sync(wsU);

    // ---------------- Phase C (block 0): log-softmax finalize, loss/state update
    if (bid == 0) {
      if (tid == 0) {
        float gm = -INFINITY, gs = 0.0f; int gi = 0x7fffffff;
        for (int b = 0; b < NBLK; ++b) {
          float m2 = wsF[WS_PART + b * 4 + 0];
          int   i2 = wsI[WS_PART + b * 4 + 2];
          if (m2 > gm) { gm = m2; gi = i2; }
          else if (m2 == gm && i2 < gi) gi = i2;
        }
        for (int b = 0; b < NBLK; ++b)
          gs += wsF[WS_PART + b * 4 + 1] * __expf(wsF[WS_PART + b * 4 + 0] - gm);
        float logp_tgt = wsF[WS_YTGT] - gm - __logf(gs);
        int done = wsI[WS_DONE];
        int active = !done;
        if (active) {
          wsF[WS_LOSS] += -logp_tgt;
          if (gi == EOS_TOK) wsI[WS_DONE] = 1;
          wsI[WS_TOK] = gi;
        }
        sh_flag = active;
      }
      __syncthreads();
      if (sh_flag) sh_h[tid] = sh_h2[tid];
      __syncthreads();
    }
  }

  if (bid == 0 && tid == 0) p.out[0] = wsF[WS_LOSS];
}

extern "C" void kernel_launch(void* const* d_in, const int* in_sizes, int n_in,
                              void* d_out, int out_size, void* d_ws, size_t ws_size,
                              hipStream_t stream) {
  Params prm;
  prm.in_tok  = (const int*)d_in[0];
  prm.in_len  = (const int*)d_in[1];
  prm.tgt_tok = (const int*)d_in[2];
  prm.tgt_len = (const int*)d_in[3];
  prm.enc_emb  = (const float*)d_in[4];
  prm.dec_emb  = (const float*)d_in[5];
  prm.enc_W_ih = (const float*)d_in[6];
  prm.enc_W_hh = (const float*)d_in[7];
  prm.enc_b_ih = (const float*)d_in[8];
  prm.enc_b_hh = (const float*)d_in[9];
  prm.dec_W_ih = (const float*)d_in[10];
  prm.dec_W_hh = (const float*)d_in[11];
  prm.dec_b_ih = (const float*)d_in[12];
  prm.dec_b_hh = (const float*)d_in[13];
  prm.attn_W   = (const float*)d_in[14];
  prm.attn_b   = (const float*)d_in[15];
  prm.comb_W   = (const float*)d_in[16];
  prm.comb_b   = (const float*)d_in[17];
  prm.out_W    = (const float*)d_in[18];
  prm.out_b    = (const float*)d_in[19];
  prm.out = (float*)d_out;
  prm.ws  = (unsigned*)d_ws;
  prm.useWbf = (ws_size >= (size_t)WS_WORDS_FULL * 4u) ? 1 : 0;

  hipLaunchKernelGGL(grumt_init, dim3(1), dim3(32), 0, stream, (unsigned*)d_ws);
  hipLaunchKernelGGL(grumt_main, dim3(NBLK), dim3(TPB), 0, stream, prm);
}
